// graph_encode_28759101014800
// MI455X (gfx1250) — compile-verified, matrix operand order checked by reading, not run on hardware
//
#include <hip/hip_runtime.h>

#define B_   2
#define E_   4096
#define R_   4096
#define N_   8192
#define M_   16
#define H_   512
#define NH_  4
#define HD_  128
#define DFF_ 2048
#define P_   2
#define BN_  (B_ * N_)            // 16384 nodes total
#define SCALE_ 0.044194173824159216f   // 1/sqrt(512)
#define EPS_ 1e-5f

typedef __bf16 bf16;
typedef __attribute__((ext_vector_type(16))) __bf16 v16bf;
typedef __attribute__((ext_vector_type(8)))  __bf16 v8bf;
typedef __attribute__((ext_vector_type(8)))  float  v8f;

// ---------------------------------------------------------------------------
// WMMA helpers (CDNA5 wave32, v_wmma_f32_16x16x32_bf16)
// ---------------------------------------------------------------------------
__device__ __forceinline__ v8f wmma_bf16(v16bf a, v16bf b, v8f c) {
  // (neg_a, A, neg_b, B, c_mod, C, reuse_a, reuse_b)
  return __builtin_amdgcn_wmma_f32_16x16x32_bf16(false, a, false, b, (short)0, c,
                                                 false, false);
}

// 16x32 bf16 fragment from a row-major [rows][pitch] array (LDS or global).
// ISA layout: lane = {row|col} = lane&15, khalf = lane>>4.
//   elements 0..7  -> K = khalf*8 + 0..7        (16 contiguous bytes)
//   elements 8..15 -> K = 16 + khalf*8 + 0..7   (16 contiguous bytes)
__device__ __forceinline__ v16bf load_frag_rm(const bf16* base, int pitch,
                                              int row, int k0, int kh) {
  const bf16* p = base + (size_t)row * pitch + k0 + kh * 8;
  v8bf lo = *(const v8bf*)p;
  v8bf hi = *(const v8bf*)(p + 16);
  return __builtin_shufflevector(lo, hi, 0, 1, 2, 3, 4, 5, 6, 7,
                                 8, 9, 10, 11, 12, 13, 14, 15);
}

// Block-wide sum over 512 threads (16 wave32 waves).
__device__ __forceinline__ float block_sum(float v, float* red, int tid) {
#pragma unroll
  for (int m = 16; m >= 1; m >>= 1) v += __shfl_xor(v, m, 32);
  if ((tid & 31) == 0) red[tid >> 5] = v;
  __syncthreads();
  float s = (tid < 16) ? red[tid] : 0.0f;
  if (tid < 32) {
#pragma unroll
    for (int m = 8; m >= 1; m >>= 1) s += __shfl_xor(s, m, 32);
    if (tid == 0) red[0] = s;
  }
  __syncthreads();
  float r = red[0];
  __syncthreads();
  return r;
}

// ---------------------------------------------------------------------------
// vg = concat(vents, vrels)  -> f32 (lives in d_out) + bf16 copy (ws)
// ---------------------------------------------------------------------------
__global__ void k_build_vg(const float* __restrict__ vents,
                           const float* __restrict__ vrels,
                           float* __restrict__ vgf, bf16* __restrict__ vgb,
                           long total) {
  long i = (long)blockIdx.x * blockDim.x + threadIdx.x;
  if (i >= total) return;
  const long per_b = (long)N_ * H_;
  const long eh = (long)E_ * H_;
  int b = (int)(i / per_b);
  long rem = i - (long)b * per_b;
  float v = (rem < eh) ? vents[(long)b * eh + rem]
                       : vrels[(long)b * ((long)R_ * H_) + (rem - eh)];
  vgf[i] = v;
  vgb[i] = (bf16)v;
}

// f32 [P][K][N] -> bf16 transposed [P][N][K]  (one-time, tiny)
__global__ void k_cvt_t(const float* __restrict__ in, bf16* __restrict__ out,
                        int K, int Ncols, long total) {
  long i = (long)blockIdx.x * blockDim.x + threadIdx.x;
  if (i >= total) return;
  long per = (long)K * Ncols;
  int p = (int)(i / per);
  long rem = i - (long)p * per;
  int k = (int)(rem / Ncols);
  int n = (int)(rem - (long)k * Ncols);
  out[(long)p * per + (long)n * K + k] = (bf16)in[i];
}

// ---------------------------------------------------------------------------
// Q = vg @ Wq : 16 rows x 512 cols per workgroup, K=512.
// 2 output tiles per wave sharing one A fragment per k-step.
// ---------------------------------------------------------------------------
__global__ __launch_bounds__(512) void k_qproj(const bf16* __restrict__ vg_bf,
                                               const bf16* __restrict__ Wqt,
                                               float* __restrict__ Qout) {
  const int row0 = blockIdx.x * 16;
  const int tid = threadIdx.x, lane = tid & 31, wave = tid >> 5;
  const int rowa = lane & 15, kh = lane >> 4, col = lane & 15;
  __shared__ __align__(16) bf16 As[16][520];
  for (int i = tid; i < 16 * 512; i += 512) {
    int m = i >> 9, d = i & 511;
    As[m][d] = vg_bf[(size_t)(row0 + m) * H_ + d];
  }
  __syncthreads();
  const int n0a = wave << 4, n0b = (wave + 16) << 4;
  v8f c0 = {}, c1 = {};
#pragma unroll 2
  for (int kk = 0; kk < 16; ++kk) {
    v16bf a = load_frag_rm(&As[0][0], 520, rowa, kk * 32, kh);
    v16bf b0 = load_frag_rm(Wqt, H_, n0a + col, kk * 32, kh);
    c0 = wmma_bf16(a, b0, c0);
    v16bf b1 = load_frag_rm(Wqt, H_, n0b + col, kk * 32, kh);
    c1 = wmma_bf16(a, b1, c1);
  }
#pragma unroll
  for (int r = 0; r < 8; ++r) {
    size_t go = (size_t)(row0 + r + 8 * kh) * H_;
    Qout[go + n0a + col] = c0[r];
    Qout[go + n0b + col] = c1[r];
  }
}

// ---------------------------------------------------------------------------
// Fused per-node kernel: gather neighbors -> K/V projection (WMMA, the
// dominant 550 GF of the model) -> masked softmax attention -> LayerNorm.
// One workgroup (16 waves) per node; 1024 WMMAs per node. Each wave owns
// 2 K tiles + 2 V tiles and reuses one A fragment across all 4 per k-step.
// ---------------------------------------------------------------------------
__global__ __launch_bounds__(512) void k_attn(
    const bf16* __restrict__ vg_bf, const int* __restrict__ adj,
    const int* __restrict__ lens, const float* __restrict__ Qbuf,
    const bf16* __restrict__ Wkt, const bf16* __restrict__ Wvt,
    const float* __restrict__ g1, const float* __restrict__ be1,
    float* __restrict__ Tout, bf16* __restrict__ Tbf_out) {
  const int node = blockIdx.x;  // b*N + n
  const int tid = threadIdx.x, lane = tid & 31, wave = tid >> 5;
  const int rowa = lane & 15, kh = lane >> 4, col = lane & 15;

  __shared__ __align__(16) bf16 ng[16][520];   // gathered neighbors (bf16)
  __shared__ __align__(16) bf16 Kl[16][520];   // K tile
  __shared__ __align__(16) bf16 Vl[16][520];   // V tile
  __shared__ float Qrow[512];
  __shared__ float sc[4][16];
  __shared__ float arow[4][16];
  __shared__ float attnrow[512];
  __shared__ float red[16];
  __shared__ int adjs[16];
  __shared__ int slen;

  if (tid < 16) adjs[tid] = adj[(size_t)node * M_ + tid];
  if (tid == 0) slen = lens[node];
  Qrow[tid] = Qbuf[(size_t)node * H_ + tid];
  __syncthreads();
  const int len = slen;
  const int bbase = (node / N_) * N_;
  // stage masked neighbor matrix (datamask: m < len)
  for (int i = tid; i < 16 * 512; i += 512) {
    int m = i >> 9, d = i & 511;
    bf16 v = (bf16)0.0f;
    if (m < len) v = vg_bf[(size_t)(bbase + adjs[m]) * H_ + d];
    ng[m][d] = v;
  }
  __syncthreads();

  // K/V projection: per wave 4 accumulators (2 K-column tiles, 2 V-column
  // tiles) share one A fragment per k-step -> 1 ds-frag feeds 4 WMMAs.
  {
    const int nA = wave << 4;
    const int nB = (wave + 16) << 4;
    v8f ck0 = {}, ck1 = {}, cv0 = {}, cv1 = {};
#pragma unroll 2
    for (int kk = 0; kk < 16; ++kk) {
      v16bf a = load_frag_rm(&ng[0][0], 520, rowa, kk * 32, kh);
      v16bf bk0 = load_frag_rm(Wkt, H_, nA + col, kk * 32, kh);
      ck0 = wmma_bf16(a, bk0, ck0);
      v16bf bk1 = load_frag_rm(Wkt, H_, nB + col, kk * 32, kh);
      ck1 = wmma_bf16(a, bk1, ck1);
      v16bf bv0 = load_frag_rm(Wvt, H_, nA + col, kk * 32, kh);
      cv0 = wmma_bf16(a, bv0, cv0);
      v16bf bv1 = load_frag_rm(Wvt, H_, nB + col, kk * 32, kh);
      cv1 = wmma_bf16(a, bv1, cv1);
    }
#pragma unroll
    for (int r = 0; r < 8; ++r) {
      int m = r + 8 * kh;
      Kl[m][nA + col] = (bf16)ck0[r];
      Kl[m][nB + col] = (bf16)ck1[r];
      Vl[m][nA + col] = (bf16)cv0[r];
      Vl[m][nB + col] = (bf16)cv1[r];
    }
  }
  __syncthreads();

  // scores (NH=4 heads x M=16 neighbors), attnmask: m <= len
  if (tid < 64) {
    int h = tid >> 4, m = tid & 15;
    float s;
    if (m <= len) {
      float acc = 0.f;
      const bf16* kr = &Kl[m][h * HD_];
      const float* qr = &Qrow[h * HD_];
#pragma unroll 8
      for (int d = 0; d < HD_; ++d) acc += qr[d] * (float)kr[d];
      s = acc * SCALE_;
    } else {
      s = -__builtin_inff();
    }
    sc[h][m] = s;
  }
  __syncthreads();
  if (tid < 4) {
    const int h = tid;
    float mx = -__builtin_inff();
#pragma unroll
    for (int m = 0; m < 16; ++m) mx = fmaxf(mx, sc[h][m]);
    float sum = 0.f, ee[16];
#pragma unroll
    for (int m = 0; m < 16; ++m) {
      float v = __expf(sc[h][m] - mx);
      ee[m] = v;
      sum += v;
    }
    float inv = 1.f / sum;
#pragma unroll
    for (int m = 0; m < 16; ++m) arow[h][m] = ee[m] * inv;
  }
  __syncthreads();
  {
    const int d = tid, h = d >> 7;
    float acc = 0.f;
#pragma unroll
    for (int m = 0; m < 16; ++m) acc += arow[h][m] * (float)Vl[m][d];
    attnrow[d] = acc;
  }
  __syncthreads();
  {
    float x = attnrow[tid];
    float s1 = block_sum(x, red, tid);
    float s2 = block_sum(x * x, red, tid);
    float mu = s1 * (1.f / H_);
    float var = s2 * (1.f / H_) - mu * mu;
    float rs = rsqrtf(var + EPS_);
    float tv = (x - mu) * rs * g1[tid] + be1[tid];
    Tout[(size_t)node * H_ + tid] = tv;
    Tbf_out[(size_t)node * H_ + tid] = (bf16)tv;
  }
}

// ---------------------------------------------------------------------------
// FFN1: hact = leaky(t @ W1 + b1), 16 rows x 2048 cols per WG, K=512.
// 8 tiles per wave as 2 passes of 4 accumulators sharing one A fragment.
// ---------------------------------------------------------------------------
__global__ __launch_bounds__(512) void k_ffn1(const bf16* __restrict__ Tbf,
                                              const bf16* __restrict__ W1t,
                                              const float* __restrict__ b1,
                                              const float* __restrict__ a1,
                                              bf16* __restrict__ Hact) {
  const int row0 = blockIdx.x * 16;
  const int tid = threadIdx.x, lane = tid & 31, wave = tid >> 5;
  const int rowa = lane & 15, kh = lane >> 4, col = lane & 15;
  __shared__ __align__(16) bf16 As[16][520];
  for (int i = tid; i < 16 * 512; i += 512) {
    int m = i >> 9, d = i & 511;
    As[m][d] = Tbf[(size_t)(row0 + m) * H_ + d];
  }
  __syncthreads();
  for (int pass = 0; pass < 2; ++pass) {
    const int jbase = pass * 64 + wave;
    int n0[4];
#pragma unroll
    for (int t = 0; t < 4; ++t) n0[t] = (jbase + 16 * t) << 4;
    v8f c[4] = {{}, {}, {}, {}};
#pragma unroll 2
    for (int kk = 0; kk < 16; ++kk) {
      v16bf a = load_frag_rm(&As[0][0], 520, rowa, kk * 32, kh);
#pragma unroll
      for (int t = 0; t < 4; ++t) {
        v16bf b = load_frag_rm(W1t, H_, n0[t] + col, kk * 32, kh);
        c[t] = wmma_bf16(a, b, c[t]);
      }
    }
#pragma unroll
    for (int t = 0; t < 4; ++t) {
      const float bb = b1[n0[t] + col], al = a1[n0[t] + col];
#pragma unroll
      for (int r = 0; r < 8; ++r) {
        int m = r + 8 * kh;
        float x = c[t][r] + bb;
        x = (x >= 0.f) ? x : al * x;
        Hact[(size_t)(row0 + m) * DFF_ + n0[t] + col] = (bf16)x;
      }
    }
  }
}

// ---------------------------------------------------------------------------
// FFN2 + residual + LayerNorm: 16 rows x full 512 cols per WG, K=2048.
// A staged in 4 LDS chunks of 16x512; the next chunk is prefetched
// (global_prefetch_b8) while the current chunk's WMMAs run.
// ---------------------------------------------------------------------------
__global__ __launch_bounds__(512) void k_ffn2_ln(
    const bf16* __restrict__ Hact, const bf16* __restrict__ W2t,
    const float* __restrict__ b2, const float* __restrict__ Tres,
    const float* __restrict__ g1, const float* __restrict__ be1,
    float* __restrict__ vg_out, bf16* __restrict__ vg_bf_out) {
  const int row0 = blockIdx.x * 16;
  const int tid = threadIdx.x, lane = tid & 31, wave = tid >> 5;
  const int rowa = lane & 15, kh = lane >> 4, col = lane & 15;
  __shared__ __align__(16) bf16 As[16][520];
  __shared__ __align__(16) float xbuf[16][516];
  v8f c0 = {}, c1 = {};
  const int n0a = wave << 4, n0b = (wave + 16) << 4;
  for (int chunk = 0; chunk < 4; ++chunk) {
    for (int i = tid; i < 16 * 512; i += 512) {
      int m = i >> 9, d = i & 511;
      As[m][d] = Hact[(size_t)(row0 + m) * DFF_ + chunk * 512 + d];
    }
    __syncthreads();
    if (chunk < 3) {
      // Prefetch the next A chunk (16 rows x 1KB) while this chunk computes.
      const bf16* pf = Hact + (size_t)(row0 + (tid >> 5)) * DFF_ +
                       (chunk + 1) * 512 + (size_t)(tid & 31) * 16;
      __builtin_prefetch(pf, 0, 0);
    }
#pragma unroll 2
    for (int kk = 0; kk < 16; ++kk) {
      v16bf a = load_frag_rm(&As[0][0], 520, rowa, kk * 32, kh);
      const int k0g = chunk * 512 + kk * 32;
      v16bf ba = load_frag_rm(W2t, DFF_, n0a + col, k0g, kh);
      c0 = wmma_bf16(a, ba, c0);
      v16bf bb = load_frag_rm(W2t, DFF_, n0b + col, k0g, kh);
      c1 = wmma_bf16(a, bb, c1);
    }
    __syncthreads();
  }
  {
    const float ba = b2[n0a + col], bb = b2[n0b + col];
#pragma unroll
    for (int r = 0; r < 8; ++r) {
      int m = r + 8 * kh;
      size_t go = (size_t)(row0 + m) * H_;
      xbuf[m][n0a + col] = c0[r] + ba + Tres[go + n0a + col];
      xbuf[m][n0b + col] = c1[r] + bb + Tres[go + n0b + col];
    }
  }
  __syncthreads();
  {
    const int m = wave;  // one wave per row for the LayerNorm
    float s = 0.f, s2 = 0.f;
    for (int d = lane; d < H_; d += 32) {
      float v = xbuf[m][d];
      s += v;
      s2 += v * v;
    }
#pragma unroll
    for (int k = 16; k >= 1; k >>= 1) {
      s += __shfl_xor(s, k, 32);
      s2 += __shfl_xor(s2, k, 32);
    }
    float mu = s * (1.f / H_);
    float var = s2 * (1.f / H_) - mu * mu;
    float rs = rsqrtf(var + EPS_);
    for (int d = lane; d < H_; d += 32) {
      float v = (xbuf[m][d] - mu) * rs * g1[d] + be1[d];
      size_t o = (size_t)(row0 + m) * H_ + d;
      vg_out[o] = v;
      vg_bf_out[o] = (bf16)v;
    }
  }
}

__global__ void k_finalize(const float* __restrict__ vgf,
                           float* __restrict__ glob,
                           float* __restrict__ emask) {
  int i = blockIdx.x * blockDim.x + threadIdx.x;
  if (i < B_ * H_) {
    int b = i / H_, d = i % H_;
    glob[i] = vgf[((size_t)b * N_ + E_) * H_ + d];
  }
  if (i < BN_) emask[i] = 1.0f;
}

// ---------------------------------------------------------------------------
extern "C" void kernel_launch(void* const* d_in, const int* in_sizes, int n_in,
                              void* d_out, int out_size, void* d_ws,
                              size_t ws_size, hipStream_t stream) {
  (void)in_sizes; (void)n_in; (void)out_size; (void)ws_size;
  const float* vents = (const float*)d_in[0];
  const float* vrels = (const float*)d_in[1];
  const int* adj = (const int*)d_in[2];
  const int* lens = (const int*)d_in[3];
  const float* Wq = (const float*)d_in[4];
  const float* Wk = (const float*)d_in[5];
  const float* Wv = (const float*)d_in[6];
  const float* W1 = (const float*)d_in[7];
  const float* b1 = (const float*)d_in[8];
  const float* W2 = (const float*)d_in[9];
  const float* b2 = (const float*)d_in[10];
  const float* g1 = (const float*)d_in[11];
  const float* be1 = (const float*)d_in[12];
  const float* a1 = (const float*)d_in[13];

  // workspace layout (~180 MB)
  char* ws = (char*)d_ws;
  size_t off = 0;
  auto alloc = [&](size_t bytes) {
    char* p = ws + off;
    off += (bytes + 255) & ~(size_t)255;
    return p;
  };
  bf16* Wqt = (bf16*)alloc((size_t)P_ * H_ * H_ * sizeof(bf16));
  bf16* Wkt = (bf16*)alloc((size_t)P_ * H_ * H_ * sizeof(bf16));
  bf16* Wvt = (bf16*)alloc((size_t)P_ * H_ * H_ * sizeof(bf16));
  bf16* W1t = (bf16*)alloc((size_t)P_ * H_ * DFF_ * sizeof(bf16));
  bf16* W2t = (bf16*)alloc((size_t)P_ * DFF_ * H_ * sizeof(bf16));
  bf16* vg_bf = (bf16*)alloc((size_t)BN_ * H_ * sizeof(bf16));
  float* Qf = (float*)alloc((size_t)BN_ * H_ * sizeof(float));
  float* Tf = (float*)alloc((size_t)BN_ * H_ * sizeof(float));
  bf16* Tbf = (bf16*)alloc((size_t)BN_ * H_ * sizeof(bf16));
  bf16* Hact = (bf16*)alloc((size_t)BN_ * DFF_ * sizeof(bf16));

  // d_out = glob (B*H) | vg (B*N*H) | emask (B*N), all f32
  float* glob = (float*)d_out;
  float* vgf = glob + (size_t)B_ * H_;
  float* emask = vgf + (size_t)BN_ * H_;

  {
    long total = (long)BN_ * H_;
    k_build_vg<<<dim3((unsigned)((total + 255) / 256)), dim3(256), 0, stream>>>(
        vents, vrels, vgf, vg_bf, total);
  }
  {
    long t3 = (long)P_ * H_ * H_;
    k_cvt_t<<<dim3((unsigned)((t3 + 255) / 256)), dim3(256), 0, stream>>>(Wq, Wqt, H_, H_, t3);
    k_cvt_t<<<dim3((unsigned)((t3 + 255) / 256)), dim3(256), 0, stream>>>(Wk, Wkt, H_, H_, t3);
    k_cvt_t<<<dim3((unsigned)((t3 + 255) / 256)), dim3(256), 0, stream>>>(Wv, Wvt, H_, H_, t3);
    long tf = (long)P_ * H_ * DFF_;
    k_cvt_t<<<dim3((unsigned)((tf + 255) / 256)), dim3(256), 0, stream>>>(W1, W1t, H_, DFF_, tf);
    k_cvt_t<<<dim3((unsigned)((tf + 255) / 256)), dim3(256), 0, stream>>>(W2, W2t, DFF_, H_, tf);
  }

  for (int p = 0; p < P_; ++p) {
    const bf16* wq = Wqt + (size_t)p * H_ * H_;
    const bf16* wk = Wkt + (size_t)p * H_ * H_;
    const bf16* wv = Wvt + (size_t)p * H_ * H_;
    const bf16* w1 = W1t + (size_t)p * H_ * DFF_;
    const bf16* w2 = W2t + (size_t)p * DFF_ * H_;
    const float* g1p = g1 + (size_t)p * H_;
    const float* be1p = be1 + (size_t)p * H_;
    const float* b1p = b1 + (size_t)p * DFF_;
    const float* b2p = b2 + (size_t)p * H_;
    const float* a1p = a1 + (size_t)p * DFF_;

    k_qproj<<<dim3(BN_ / 16), dim3(512), 0, stream>>>(vg_bf, wq, Qf);
    k_attn<<<dim3(BN_), dim3(512), 0, stream>>>(vg_bf, adj, lens, Qf, wk, wv,
                                                g1p, be1p, Tf, Tbf);
    k_ffn1<<<dim3(BN_ / 16), dim3(512), 0, stream>>>(Tbf, w1, b1p, a1p, Hact);
    k_ffn2_ln<<<dim3(BN_ / 16), dim3(512), 0, stream>>>(Hact, w2, b2p, Tf, g1p,
                                                        be1p, vgf, vg_bf);
  }
  k_finalize<<<dim3((BN_ + 255) / 256), dim3(256), 0, stream>>>(vgf, glob, emask);
}
